// RAFT_77661598646586
// MI455X (gfx1250) — matrix operand order repaired and checked by reading, not compiled, not run on hardware
//
#include <hip/hip_runtime.h>

// ---------------------------------------------------------------------------
// Model constants
// ---------------------------------------------------------------------------
#define D_MODEL 256
#define N_HEADS 8
#define D_HEAD  32
#define N_LEVELS 6
#define N_POINTS 4
#define D_FFN   1024
#define N_ENC   6
#define N_QUERY 25
#define L_TOTAL 3360
#define BATCH   2
#define BL      (BATCH * L_TOTAL)     // 6720
#define BQ      (BATCH * N_QUERY)     // 50
#define H0      32
#define W0      40
#define I_H     256
#define I_W     320

// ---------------------------------------------------------------------------
// TDM availability (toolchain-gated; falls back to VGPR staging)
// ---------------------------------------------------------------------------
#ifdef __has_builtin
#if __has_builtin(__builtin_amdgcn_tensor_load_to_lds) && \
    __has_builtin(__builtin_amdgcn_s_wait_tensorcnt)
#define HAVE_TDM 1
#endif
#endif
#ifndef HAVE_TDM
#define HAVE_TDM 0
#endif

// ---------------------------------------------------------------------------
// WMMA types (bf16 in, f32 accumulate)  -- 16x16x32 per ISA 7.12
// ---------------------------------------------------------------------------
typedef __attribute__((ext_vector_type(16))) __bf16 v16bf;
typedef __attribute__((ext_vector_type(8)))  float  v8f;

union Frag16 { v16bf v; uint4 q[2]; unsigned int u[8]; unsigned short h[16]; };
union Acc8   { v8f v; float f[8]; };

__device__ __forceinline__ unsigned short f2bf(float f) {
    unsigned int u = __builtin_bit_cast(unsigned int, f);
    unsigned int r = u + 0x7FFFu + ((u >> 16) & 1u);   // RNE
    return (unsigned short)(r >> 16);
}
__device__ __forceinline__ unsigned int pack2bf(float a, float b) {
    return (unsigned int)f2bf(a) | ((unsigned int)f2bf(b) << 16);
}

__device__ __forceinline__ float gelu_f(float x) {
    return 0.5f * x * (1.f + tanhf(0.7978845608f * (x + 0.044715f * x * x * x)));
}

__device__ __forceinline__ float waveSum(float v) {
    #pragma unroll
    for (int m = 16; m >= 1; m >>= 1) v += __shfl_xor(v, m, 32);
    return v;
}

#if HAVE_TDM
typedef unsigned int tdm_v4u __attribute__((ext_vector_type(4)));
typedef int          tdm_v8i __attribute__((ext_vector_type(8)));
typedef int          tdm_v4i __attribute__((ext_vector_type(4)));

// TDM: async-load a 64-row x 32-col fp32 tile (row stride = strideElems) into
// LDS at ldsOff, inserting 4 dwords of padding per 128B row (-> 144B LDS row
// stride).  D# layout per CDNA5 ISA.
__device__ __forceinline__ void tdm_load_tile_f32(const float* gsrc,
                                                  unsigned ldsOff,
                                                  int strideElems)
{
    unsigned long long ga = (unsigned long long)gsrc;
    tdm_v4u g0;
    g0[0] = 1u;                                           // count=1, user mode
    g0[1] = ldsOff;                                       // lds_addr (bytes)
    g0[2] = (unsigned)(ga & 0xFFFFFFFFu);                 // global_addr lo
    g0[3] = (unsigned)((ga >> 32) & 0x01FFFFFFu)          // global_addr hi
            | (2u << 30);                                 // type = 2 (image)
    tdm_v8i g1;
    g1[0] = (2 << 16)        // data_size = 4B
          | (1 << 20)        // pad_enable
          | (4 << 22)        // pad_interval code 4 -> every 32 dwords (128B)
          | (3 << 25);       // pad_amount  code 3 -> 4 dwords (16B)
    g1[1] = (32 << 16);      // tensor_dim0 = 32 (low 16 bits at [31:16])
    g1[2] = (64 << 16);      // tensor_dim0 hi = 0 ; tensor_dim1 = 64 (low 16)
    g1[3] = (32 << 16);      // tensor_dim1 hi = 0 ; tile_dim0 = 32
    g1[4] = 64;              // tile_dim1 = 64 ; tile_dim2 = 0
    g1[5] = strideElems;     // tensor_dim0_stride (low 32, data_size units)
    g1[6] = 0;               // stride hi ; tensor_dim1_stride lo
    g1[7] = 0;
    tdm_v4i gz = {0, 0, 0, 0};
#if __clang_major__ >= 23
    tdm_v8i gz8 = {0, 0, 0, 0, 0, 0, 0, 0};
    __builtin_amdgcn_tensor_load_to_lds(g0, g1, gz, gz, gz8, 0);
#else
    __builtin_amdgcn_tensor_load_to_lds(g0, g1, gz, gz, 0);
#endif
}
#endif // HAVE_TDM

// ---------------------------------------------------------------------------
// WMMA GEMM:  C[M,N] = act( (A (+ A2 broadcast-by-rows, period p2)) @ W + bias )
//   HASA2  : fuse elementwise A2 add (row-broadcast, <=1 period wrap)
//   BOUNDS : row/col guards (tiny decoder GEMMs only)
// Hot instances (!BOUNDS,!HASA2) stage the A tile with the Tensor Data Mover
// (raw fp32, bf16 conversion folded into fragment build); other instances use
// vectorized VGPR staging.  B tile stored transposed so each fragment is two
// 16-byte LDS loads.  4 waves/block, 64x64 tile, 2x2 v_wmma_f32_16x16x32_bf16.
// ---------------------------------------------------------------------------
#define TM 64
#define TN 64
#define TK 32
#define LSTR 40    // bf16 LDS row stride (ushorts): 80B
#define FSTR 36    // fp32 TDM LDS row stride (floats): 144B

template <bool HASA2, bool BOUNDS>
__global__ void __launch_bounds__(128)
gemm_wmma(const float* __restrict__ A,
          const float* __restrict__ A2, int p2,
          const float* __restrict__ Wt, const float* __restrict__ bias,
          float* __restrict__ C, int M, int N, int K, int act)
{
    __shared__ __align__(16) char sAbuf[TM * FSTR * 4];   // 9216B, aliased
    __shared__ unsigned short sBT[TN][LSTR];              // B tile transposed

    float (*sAraw)[FSTR]        = (float (*)[FSTR])sAbuf;          // TDM path
    unsigned short (*sA)[LSTR]  = (unsigned short (*)[LSTR])sAbuf; // bf16 path

#if HAVE_TDM
    constexpr bool USETDM = (!BOUNDS) && (!HASA2);
#else
    constexpr bool USETDM = false;
#endif

    const int tid  = threadIdx.x;
    const int lane = tid & 31;
    const int wave = tid >> 5;
    const int wm = (wave >> 1) * 32;
    const int wn = (wave & 1) * 32;
    const int bm = blockIdx.y * TM;
    const int bn = blockIdx.x * TN;
    const int hi = lane >> 4;
    const int l15 = lane & 15;

    Acc8 acc[2][2];
    #pragma unroll
    for (int i = 0; i < 2; ++i)
        #pragma unroll
        for (int j = 0; j < 2; ++j)
            #pragma unroll
            for (int r = 0; r < 8; ++r) acc[i][j].f[r] = 0.f;

    for (int kb = 0; kb < K; kb += TK) {
        // ---------------- A tile staging ----------------
        if (USETDM) {
#if HAVE_TDM
            if (wave == 0) {
                unsigned ldsOff = (unsigned)(unsigned long long)(void*)&sAraw[0][0];
                tdm_load_tile_f32(A + (long)bm * K + kb, ldsOff, K);
            }
#endif
        } else if (!BOUNDS) {
            #pragma unroll
            for (int it = 0; it < 4; ++it) {
                int f = it * 128 + tid;
                int r = f >> 3, c4 = (f & 7) * 4;
                int gr = bm + r;
                float4 v4 = *(const float4*)(A + (long)gr * K + kb + c4);
                if (HASA2) {
                    int g2 = (gr >= p2) ? gr - p2 : gr;
                    float4 w4 = *(const float4*)(A2 + (long)g2 * K + kb + c4);
                    v4.x += w4.x; v4.y += w4.y; v4.z += w4.z; v4.w += w4.w;
                }
                *(uint2*)&sA[r][c4] =
                    make_uint2(pack2bf(v4.x, v4.y), pack2bf(v4.z, v4.w));
            }
        } else {
            #pragma unroll
            for (int it = 0; it < 16; ++it) {
                int idx = it * 128 + tid;
                int r = idx >> 5, c = idx & 31;
                int gr = bm + r;
                float val = 0.f;
                if (gr < M) {
                    val = A[(long)gr * K + kb + c];
                    if (HASA2) {
                        int g2 = (gr >= p2) ? gr - p2 : gr;
                        val += A2[(long)g2 * K + kb + c];
                    }
                }
                sA[r][c] = f2bf(val);
            }
        }
        // ---------------- B tile staging (transposed) ----------------
        if (!BOUNDS) {
            #pragma unroll
            for (int it = 0; it < 4; ++it) {
                int f = it * 128 + tid;
                int r = f >> 4, c4 = (f & 15) * 4;
                float4 v4 = *(const float4*)(Wt + (long)(kb + r) * N + bn + c4);
                sBT[c4 + 0][r] = f2bf(v4.x);
                sBT[c4 + 1][r] = f2bf(v4.y);
                sBT[c4 + 2][r] = f2bf(v4.z);
                sBT[c4 + 3][r] = f2bf(v4.w);
            }
        } else {
            #pragma unroll
            for (int it = 0; it < 16; ++it) {
                int idx = it * 128 + tid;
                int r = idx >> 6, c = idx & 63;
                int gc = bn + c;
                float val = (gc < N) ? Wt[(long)(kb + r) * N + gc] : 0.f;
                sBT[c][r] = f2bf(val);
            }
        }
        // prefetch next K slab of W into L2 (global_prefetch_b8, speculative)
        if (kb + TK < K) {
            long pg = (long)(kb + TK + (tid >> 2)) * N + bn + ((tid & 3) << 4);
            __builtin_prefetch(Wt + pg, 0, 0);
        }
#if HAVE_TDM
        if (USETDM && wave == 0) __builtin_amdgcn_s_wait_tensorcnt(0);
#endif
        __syncthreads();

        // ---------------- fragments ----------------
        Frag16 fa[2], fb[2];
        #pragma unroll
        for (int t = 0; t < 2; ++t) {
            int row = wm + t * 16 + l15;
            if (USETDM) {
                // fp32 tile -> bf16 fragment (conversion co-executes with WMMA)
                const float* pr = &sAraw[row][0];
                float4 a0 = *(const float4*)(pr + hi * 8);
                float4 a1 = *(const float4*)(pr + hi * 8 + 4);
                float4 a2 = *(const float4*)(pr + 16 + hi * 8);
                float4 a3 = *(const float4*)(pr + 16 + hi * 8 + 4);
                fa[t].u[0] = pack2bf(a0.x, a0.y);
                fa[t].u[1] = pack2bf(a0.z, a0.w);
                fa[t].u[2] = pack2bf(a1.x, a1.y);
                fa[t].u[3] = pack2bf(a1.z, a1.w);
                fa[t].u[4] = pack2bf(a2.x, a2.y);
                fa[t].u[5] = pack2bf(a2.z, a2.w);
                fa[t].u[6] = pack2bf(a3.x, a3.y);
                fa[t].u[7] = pack2bf(a3.z, a3.w);
            } else {
                const unsigned short* pr = &sA[row][0];
                fa[t].q[0] = *(const uint4*)(pr + hi * 8);
                fa[t].q[1] = *(const uint4*)(pr + 16 + hi * 8);
            }
        }
        #pragma unroll
        for (int t = 0; t < 2; ++t) {
            const unsigned short* pc = &sBT[wn + t * 16 + l15][0];
            fb[t].q[0] = *(const uint4*)(pc + hi * 8);
            fb[t].q[1] = *(const uint4*)(pc + 16 + hi * 8);
        }
        #pragma unroll
        for (int i = 0; i < 2; ++i)
            #pragma unroll
            for (int j = 0; j < 2; ++j)
                acc[i][j].v = __builtin_amdgcn_wmma_f32_16x16x32_bf16(
                    false, fa[i].v, false, fb[j].v,
                    (short)0, acc[i][j].v, false, false);
        __syncthreads();
    }

    // ---------------- epilogue ----------------
    #pragma unroll
    for (int i = 0; i < 2; ++i) {
        #pragma unroll
        for (int j = 0; j < 2; ++j) {
            int cbase = bn + wn + j * 16 + l15;
            if (BOUNDS && cbase >= N) continue;
            float bval = bias ? bias[cbase] : 0.f;
            #pragma unroll
            for (int r = 0; r < 8; ++r) {
                int row = bm + wm + i * 16 + r + (hi ? 8 : 0);
                if (!BOUNDS || row < M) {
                    float o = acc[i][j].f[r] + bval;
                    if (act == 1) o = gelu_f(o);
                    C[(long)row * N + cbase] = o;
                }
            }
        }
    }
}

// ---------------------------------------------------------------------------
// y[row,col] = a[rowmod(pa)] (+ b[rowmod(pb)])  rows of 256, <=1 period wrap
// ---------------------------------------------------------------------------
__global__ void bcast_add(const float* __restrict__ a, int pa,
                          const float* __restrict__ b, int pb,
                          float* __restrict__ y, long n)
{
    long i = (long)blockIdx.x * blockDim.x + threadIdx.x;
    if (i >= n) return;
    int row = (int)(i >> 8);
    int col = (int)(i & 255);
    int ra = (row >= pa) ? row - pa : row;
    float v = a[(long)ra * 256 + col];
    if (b) {
        int rb = (row >= pb) ? row - pb : row;
        v += b[(long)rb * 256 + col];
    }
    y[i] = v;
}

// ---------------------------------------------------------------------------
// y = LayerNorm(x + r) * g + b   over D_MODEL=256, one wave32 per token
// ---------------------------------------------------------------------------
__global__ void add_ln(const float* __restrict__ x, const float* __restrict__ r,
                       const float* __restrict__ g, const float* __restrict__ b,
                       float* __restrict__ y, int ntok)
{
    int gid = blockIdx.x * blockDim.x + threadIdx.x;
    int wave = gid >> 5, lane = gid & 31;
    if (wave >= ntok) return;
    long base = (long)wave * D_MODEL;
    float vals[8];
    float s = 0.f;
    #pragma unroll
    for (int j = 0; j < 8; ++j) {
        float v = x[base + lane + 32 * j] + r[base + lane + 32 * j];
        vals[j] = v; s += v;
    }
    s = waveSum(s);
    float mu = s * (1.f / 256.f);
    float q = 0.f;
    #pragma unroll
    for (int j = 0; j < 8; ++j) { float d = vals[j] - mu; q += d * d; }
    q = waveSum(q) * (1.f / 256.f);
    float inv = rsqrtf(q + 1e-5f);
    #pragma unroll
    for (int j = 0; j < 8; ++j) {
        int c = lane + 32 * j;
        y[base + c] = (vals[j] - mu) * inv * g[c] + b[c];
    }
}

// ---------------------------------------------------------------------------
// in-place softmax over contiguous groups of 24 (attention weights)
// ---------------------------------------------------------------------------
__global__ void softmax24(float* __restrict__ aw, long ngroups)
{
    long t = (long)blockIdx.x * blockDim.x + threadIdx.x;
    if (t >= ngroups) return;
    float* p = aw + t * 24;
    float mx = -1e30f;
    float e[24];
    #pragma unroll
    for (int i = 0; i < 24; ++i) { e[i] = p[i]; mx = fmaxf(mx, e[i]); }
    float s = 0.f;
    #pragma unroll
    for (int i = 0; i < 24; ++i) { e[i] = __expf(e[i] - mx); s += e[i]; }
    float inv = 1.f / s;
    #pragma unroll
    for (int i = 0; i < 24; ++i) p[i] = e[i] * inv;
}

// ---------------------------------------------------------------------------
// encoder reference points (L_TOTAL x 2) -- constant-divisor branch chain
// ---------------------------------------------------------------------------
__global__ void make_ref(float* __restrict__ refE)
{
    int i = blockIdx.x * blockDim.x + threadIdx.x;
    if (i >= L_TOTAL) return;
    int o, Hl, Wl;
    if      (i < 1280) { o = i;        Hl = 32; Wl = 40; }
    else if (i < 1600) { o = i - 1280; Hl = 16; Wl = 20; }
    else if (i < 1680) { o = i - 1600; Hl = 8;  Wl = 10; }
    else if (i < 2960) { o = i - 1680; Hl = 32; Wl = 40; }
    else if (i < 3280) { o = i - 2960; Hl = 16; Wl = 20; }
    else               { o = i - 3280; Hl = 8;  Wl = 10; }
    int yy, xx;
    if (Wl == 40)      { yy = o / 40; xx = o - yy * 40; }
    else if (Wl == 20) { yy = o / 20; xx = o - yy * 20; }
    else               { yy = o / 10; xx = o - yy * 10; }
    refE[i * 2 + 0] = ((float)xx + 0.5f) / (float)Wl;
    refE[i * 2 + 1] = ((float)yy + 0.5f) / (float)Hl;
}

// ---------------------------------------------------------------------------
// multi-scale deformable sampling: one wave32 per (b, q, head), lane = channel
// ---------------------------------------------------------------------------
template <int LQ>
__global__ void deform_sample(const float* __restrict__ v,
                              const float* __restrict__ ref, int refPerBatch,
                              const float* __restrict__ off,
                              const float* __restrict__ aw,
                              float* __restrict__ out)
{
    const int HH[6] = {32,16,8,32,16,8};
    const int WW[6] = {40,20,10,40,20,10};
    const int SS[6] = {0,1280,1600,1680,2960,3280};
    int gid = blockIdx.x * blockDim.x + threadIdx.x;
    int wave = gid >> 5, lane = gid & 31;
    if (wave >= BATCH * LQ * N_HEADS) return;
    int h = wave % N_HEADS;
    int q = (wave / N_HEADS) % LQ;
    int b = wave / (N_HEADS * LQ);
    long tq = (long)b * LQ + q;
    long ri = refPerBatch ? tq : (long)q;
    float rx = ref[ri * 2 + 0];
    float ry = ref[ri * 2 + 1];
    float acc = 0.f;
    #pragma unroll
    for (int l = 0; l < 6; ++l) {
        const int Hl = HH[l], Wl = WW[l], s = SS[l];
        #pragma unroll
        for (int p = 0; p < 4; ++p) {
            int oi = (h * N_LEVELS + l) * N_POINTS + p;
            float ox = off[tq * 384 + oi * 2 + 0];
            float oy = off[tq * 384 + oi * 2 + 1];
            float wgt = aw[tq * 192 + oi];
            float x = (rx + ox / (float)Wl) * (float)Wl - 0.5f;
            float y = (ry + oy / (float)Hl) * (float)Hl - 0.5f;
            float x0 = floorf(x), y0 = floorf(y);
            #pragma unroll
            for (int dy = 0; dy < 2; ++dy)
                #pragma unroll
                for (int dx = 0; dx < 2; ++dx) {
                    float xi = x0 + (float)dx, yi = y0 + (float)dy;
                    float wb = (1.f - fabsf(x - xi)) * (1.f - fabsf(y - yi));
                    if (xi >= 0.f && xi < (float)Wl && yi >= 0.f && yi < (float)Hl) {
                        int idx = s + (int)yi * Wl + (int)xi;
                        acc += wgt * wb *
                            v[((long)b * L_TOTAL + idx) * D_MODEL + h * D_HEAD + lane];
                    }
                }
        }
    }
    out[tq * D_MODEL + h * D_HEAD + lane] = acc;
}

// ---------------------------------------------------------------------------
// decoder self-attention over 25 tokens: one wave32 per (b, h, qi)
// ---------------------------------------------------------------------------
__global__ void mha25(const float* __restrict__ Q, const float* __restrict__ Kh,
                      const float* __restrict__ Vh, float* __restrict__ O)
{
    int gid = blockIdx.x * blockDim.x + threadIdx.x;
    int wave = gid >> 5, lane = gid & 31;
    if (wave >= BATCH * N_HEADS * N_QUERY) return;
    int qi = wave % N_QUERY;
    int h = (wave / N_QUERY) % N_HEADS;
    int b = wave / (N_QUERY * N_HEADS);
    long hb = (long)h * D_HEAD + lane;
    float qd = Q[((long)(b * N_QUERY + qi)) * D_MODEL + hb];
    float sc[N_QUERY];
    float mx = -1e30f;
    for (int k = 0; k < N_QUERY; ++k) {
        float p = qd * Kh[((long)(b * N_QUERY + k)) * D_MODEL + hb];
        p = waveSum(p) * 0.1767766953f;       // 1/sqrt(32)
        sc[k] = p; mx = fmaxf(mx, p);
    }
    float s = 0.f;
    for (int k = 0; k < N_QUERY; ++k) { sc[k] = __expf(sc[k] - mx); s += sc[k]; }
    float inv = 1.f / s;
    float o = 0.f;
    for (int k = 0; k < N_QUERY; ++k)
        o += sc[k] * inv * Vh[((long)(b * N_QUERY + k)) * D_MODEL + hb];
    O[((long)(b * N_QUERY + qi)) * D_MODEL + hb] = o;
}

// ---------------------------------------------------------------------------
__global__ void sigmoid_k(const float* __restrict__ x, float* __restrict__ y, long n)
{
    long i = (long)blockIdx.x * blockDim.x + threadIdx.x;
    if (i < n) y[i] = 1.f / (1.f + __expf(-x[i]));
}

// flow = ref - sigmoid(inverse_sigmoid(ref) + fe)
__global__ void flow_comb(const float* __restrict__ refp, const float* __restrict__ fe,
                          float* __restrict__ fl, long n)
{
    long i = (long)blockIdx.x * blockDim.x + threadIdx.x;
    if (i >= n) return;
    float r = refp[i];
    float rc = fminf(fmaxf(r, 0.f), 1.f);
    float inv = logf(fmaxf(rc, 1e-5f) / fmaxf(1.f - rc, 1e-5f));
    float s = 1.f / (1.f + __expf(-(inv + fe[i])));
    fl[i] = r - s;
}

// ---------------------------------------------------------------------------
// A = softmax(U1 @ ctx^T); cf = (A @ flow) * [I_W, I_H]; one wave per (b, n)
// ---------------------------------------------------------------------------
__global__ void ctx_attn(const float* __restrict__ U1, const float* __restrict__ ctx,
                         const float* __restrict__ fl, float* __restrict__ cf)
{
    int gid = blockIdx.x * blockDim.x + threadIdx.x;
    int wave = gid >> 5, lane = gid & 31;
    const int NPIX = H0 * W0;               // 1280
    if (wave >= BATCH * NPIX) return;
    int n = wave % NPIX;
    int b = wave / NPIX;
    float sc[N_QUERY];
    float mx = -1e30f;
    for (int q = 0; q < N_QUERY; ++q) {
        float d = 0.f;
        #pragma unroll
        for (int j = 0; j < 8; ++j)
            d += U1[((long)(b * NPIX + n)) * D_MODEL + lane + 32 * j] *
                 ctx[((long)(b * N_QUERY + q)) * D_MODEL + lane + 32 * j];
        d = waveSum(d);
        sc[q] = d; mx = fmaxf(mx, d);
    }
    float s = 0.f;
    for (int q = 0; q < N_QUERY; ++q) { sc[q] = __expf(sc[q] - mx); s += sc[q]; }
    float inv = 1.f / s;
    float f0 = 0.f, f1 = 0.f;
    for (int q = 0; q < N_QUERY; ++q) {
        float p = sc[q] * inv;
        f0 += p * fl[(b * N_QUERY + q) * 2 + 0];
        f1 += p * fl[(b * N_QUERY + q) * 2 + 1];
    }
    if (lane == 0) {
        cf[((long)b * 2 + 0) * NPIX + n] = f0 * (float)I_W;
        cf[((long)b * 2 + 1) * NPIX + n] = f1 * (float)I_H;
    }
}

// ---------------------------------------------------------------------------
// convex upsampling: out (B, 2, 8*H0, 8*W0)
// ---------------------------------------------------------------------------
__global__ void upsample_k(const float* __restrict__ cf, const float* __restrict__ mask,
                           float* __restrict__ out)
{
    const long total = (long)BATCH * 2 * (8 * H0) * (8 * W0);
    long t = (long)blockIdx.x * blockDim.x + threadIdx.x;
    if (t >= total) return;
    int X = (int)(t % (8 * W0));
    int Y = (int)((t / (8 * W0)) % (8 * H0));
    int c = (int)((t / ((long)(8 * W0) * (8 * H0))) % 2);
    int n = (int)(t / ((long)2 * (8 * W0) * (8 * H0)));
    int h = Y >> 3, i = Y & 7, w = X >> 3, j = X & 7;
    float m[9], mx = -1e30f;
    #pragma unroll
    for (int k = 0; k < 9; ++k) {
        m[k] = mask[(((long)n * 576 + k * 64 + i * 8 + j) * H0 + h) * W0 + w];
        mx = fmaxf(mx, m[k]);
    }
    float s = 0.f;
    #pragma unroll
    for (int k = 0; k < 9; ++k) { m[k] = __expf(m[k] - mx); s += m[k]; }
    float inv = 1.f / s;
    float acc = 0.f;
    #pragma unroll
    for (int k = 0; k < 9; ++k) {
        int ki = k / 3, kj = k % 3;
        int hh = h + ki - 1, ww = w + kj - 1;
        float pv = (hh >= 0 && hh < H0 && ww >= 0 && ww < W0)
                   ? 8.f * cf[(((long)n * 2 + c) * H0 + hh) * W0 + ww] : 0.f;
        acc += m[k] * inv * pv;
    }
    out[t] = acc;
}

// ---------------------------------------------------------------------------
// host side
// ---------------------------------------------------------------------------
extern "C" void kernel_launch(void* const* d_in, const int* in_sizes, int n_in,
                              void* d_out, int out_size, void* d_ws, size_t ws_size,
                              hipStream_t stream)
{
    (void)in_sizes; (void)n_in; (void)out_size; (void)ws_size;
    auto F = [&](int i) { return (const float*)d_in[i]; };

    const float* src     = F(0);
    const float* src_pos = F(1);
    const float* U1      = F(2);
    const float* up_mask = F(3);
    const int ENC = 4;
    const int DEC = 20;
    const int MLP = 50;

    float* ws = (float*)d_ws;
    size_t o = 0;
    auto carve = [&](size_t n) { float* p = ws + o; o += n; return p; };
    float* x    = carve((size_t)BL * D_MODEL);
    float* RA   = carve((size_t)BL * 1280);
    float* refE = carve((size_t)L_TOTAL * 2);
    float* q0   = carve((size_t)BQ * D_MODEL);
    float* qry  = carve((size_t)BQ * D_MODEL);
    float* t1   = carve((size_t)BQ * D_FFN);
    float* t2   = carve((size_t)BQ * D_MODEL);
    float* refp = carve((size_t)BQ * 2);
    float* Qh   = carve((size_t)BQ * D_MODEL);
    float* Kh   = carve((size_t)BQ * D_MODEL);
    float* Vh   = carve((size_t)BQ * D_MODEL);
    float* Ob   = carve((size_t)BQ * D_MODEL);
    float* offd = carve((size_t)BQ * 384);
    float* awd  = carve((size_t)BQ * 192);
    float* attnd= carve((size_t)BQ * D_MODEL);
    float* fe   = carve((size_t)BQ * 2);
    float* ctxv = carve((size_t)BQ * D_MODEL);
    float* flv  = carve((size_t)BQ * 2);
    float* cf   = carve((size_t)BATCH * 2 * H0 * W0);

    float* vbuf = RA;
    float* offb = RA + (size_t)BL * 256;
    float* awb  = RA + (size_t)BL * 640;
    float* attn = RA + (size_t)BL * 832;
    float* ffh  = RA;
    float* ffo  = RA + (size_t)BL * 1024;

    auto gemm = [&](const float* A, const float* A2, int p2,
                    const float* Wt, const float* bias, float* C,
                    int M, int N, int K, int act) {
        dim3 g((N + TN - 1) / TN, (M + TM - 1) / TM);
        bool bounds = (M % TM) != 0 || (N % TN) != 0;
        if (A2) {
            if (bounds) gemm_wmma<true,  true ><<<g, 128, 0, stream>>>(A, A2, p2, Wt, bias, C, M, N, K, act);
            else        gemm_wmma<true,  false><<<g, 128, 0, stream>>>(A, A2, p2, Wt, bias, C, M, N, K, act);
        } else {
            if (bounds) gemm_wmma<false, true ><<<g, 128, 0, stream>>>(A, nullptr, 0, Wt, bias, C, M, N, K, act);
            else        gemm_wmma<false, false><<<g, 128, 0, stream>>>(A, nullptr, 0, Wt, bias, C, M, N, K, act);
        }
    };
    auto elems = [](long n) { return (unsigned)((n + 255) / 256); };

    // init: x = src; encoder reference points
    bcast_add<<<elems((long)BL * D_MODEL), 256, 0, stream>>>(
        src, BL, nullptr, 1, x, (long)BL * D_MODEL);
    make_ref<<<elems(L_TOTAL), 256, 0, stream>>>(refE);

    // ---- encoder: 6 layers ----
    for (int i = 0; i < N_ENC; ++i) {
        const float* Wv  = F(ENC + 0)  + (size_t)i * 256 * 256;
        const float* bv  = F(ENC + 1)  + (size_t)i * 256;
        const float* Wof = F(ENC + 2)  + (size_t)i * 256 * 384;
        const float* bof = F(ENC + 3)  + (size_t)i * 384;
        const float* Wa  = F(ENC + 4)  + (size_t)i * 256 * 192;
        const float* ba  = F(ENC + 5)  + (size_t)i * 192;
        const float* Wo  = F(ENC + 6)  + (size_t)i * 256 * 256;
        const float* bo  = F(ENC + 7)  + (size_t)i * 256;
        const float* g1  = F(ENC + 8)  + (size_t)i * 256;
        const float* b1g = F(ENC + 9)  + (size_t)i * 256;
        const float* W1  = F(ENC + 10) + (size_t)i * 256 * 1024;
        const float* bb1 = F(ENC + 11) + (size_t)i * 1024;
        const float* W2  = F(ENC + 12) + (size_t)i * 1024 * 256;
        const float* bb2 = F(ENC + 13) + (size_t)i * 256;
        const float* g2  = F(ENC + 14) + (size_t)i * 256;
        const float* b2g = F(ENC + 15) + (size_t)i * 256;

        gemm(x, nullptr, 0, Wv, bv, vbuf, BL, 256, 256, 0);
        gemm(x, src_pos, L_TOTAL, Wof, bof, offb, BL, 384, 256, 0);
        gemm(x, src_pos, L_TOTAL, Wa, ba, awb, BL, 192, 256, 0);
        softmax24<<<elems((long)BL * N_HEADS), 256, 0, stream>>>(awb, (long)BL * N_HEADS);
        deform_sample<L_TOTAL><<<elems((long)BATCH * L_TOTAL * N_HEADS * 32), 256, 0,
                                 stream>>>(vbuf, refE, 0, offb, awb, attn);
        gemm(attn, nullptr, 0, Wo, bo, offb, BL, 256, 256, 0);
        add_ln<<<elems((long)BL * 32), 256, 0, stream>>>(x, offb, g1, b1g, x, BL);
        gemm(x, nullptr, 0, W1, bb1, ffh, BL, 1024, 256, 1);
        gemm(ffh, nullptr, 0, W2, bb2, ffo, BL, 256, 1024, 0);
        add_ln<<<elems((long)BL * 32), 256, 0, stream>>>(x, ffo, g2, b2g, x, BL);
    }

    // ---- decoder ----
    const float* dWv  = F(DEC + 0);  const float* dbv  = F(DEC + 1);
    const float* dWof = F(DEC + 2);  const float* dbof = F(DEC + 3);
    const float* dWa  = F(DEC + 4);  const float* dba  = F(DEC + 5);
    const float* dWo  = F(DEC + 6);  const float* dbo  = F(DEC + 7);
    const float* dW1  = F(DEC + 10); const float* db1  = F(DEC + 11);
    const float* dW2  = F(DEC + 12); const float* db2  = F(DEC + 13);
    const float* g2d  = F(DEC + 14); const float* b2d  = F(DEC + 15);
    const float* Wq   = F(DEC + 16); const float* bq   = F(DEC + 17);
    const float* Wk   = F(DEC + 18); const float* bk   = F(DEC + 19);
    const float* Wvs  = F(DEC + 20); const float* bvs  = F(DEC + 21);
    const float* Wos  = F(DEC + 22); const float* bos  = F(DEC + 23);
    const float* gsa  = F(DEC + 24); const float* bsa  = F(DEC + 25);
    const float* gca  = F(DEC + 26); const float* bca  = F(DEC + 27);
    const float* query = F(DEC + 28);
    const float* qpos  = F(DEC + 29);

    const long fullQ = (long)BQ * D_MODEL;
    bcast_add<<<elems(fullQ), 256, 0, stream>>>(query, N_QUERY, qpos, N_QUERY, q0, fullQ);
    bcast_add<<<elems(fullQ), 256, 0, stream>>>(query, N_QUERY, nullptr, 1, qry, fullQ);

    // ref_pts = sigmoid(MLP_ref(q0))
    gemm(q0, nullptr, 0, F(MLP + 0), F(MLP + 1), t1, BQ, 256, 256, 1);
    gemm(t1, nullptr, 0, F(MLP + 2), F(MLP + 3), t2, BQ, 256, 256, 1);
    gemm(t2, nullptr, 0, F(MLP + 4), F(MLP + 5), refp, BQ, 2, 256, 0);
    sigmoid_k<<<elems(BQ * 2), 256, 0, stream>>>(refp, refp, BQ * 2);

    // self-attention
    gemm(q0, nullptr, 0, Wq, bq, Qh, BQ, 256, 256, 0);
    gemm(q0, nullptr, 0, Wk, bk, Kh, BQ, 256, 256, 0);
    gemm(qry, nullptr, 0, Wvs, bvs, Vh, BQ, 256, 256, 0);
    mha25<<<elems((long)BATCH * N_HEADS * N_QUERY * 32), 256, 0, stream>>>(Qh, Kh, Vh, Ob);
    gemm(Ob, nullptr, 0, Wos, bos, t2, BQ, 256, 256, 0);
    add_ln<<<elems((long)BQ * 32), 256, 0, stream>>>(qry, t2, gsa, bsa, qry, BQ);

    // cross deformable attention (value = encoder output x)
    gemm(x, nullptr, 0, dWv, dbv, vbuf, BL, 256, 256, 0);
    gemm(qry, qpos, N_QUERY, dWof, dbof, offd, BQ, 384, 256, 0);
    gemm(qry, qpos, N_QUERY, dWa, dba, awd, BQ, 192, 256, 0);
    softmax24<<<elems((long)BQ * N_HEADS), 256, 0, stream>>>(awd, (long)BQ * N_HEADS);
    deform_sample<N_QUERY><<<elems((long)BATCH * N_QUERY * N_HEADS * 32), 256, 0,
                             stream>>>(vbuf, refp, 1, offd, awd, attnd);
    gemm(attnd, nullptr, 0, dWo, dbo, t2, BQ, 256, 256, 0);
    add_ln<<<elems((long)BQ * 32), 256, 0, stream>>>(qry, t2, gca, bca, qry, BQ);

    // decoder FFN
    gemm(qry, nullptr, 0, dW1, db1, t1, BQ, 1024, 256, 1);
    gemm(t1, nullptr, 0, dW2, db2, t2, BQ, 256, 1024, 0);
    add_ln<<<elems((long)BQ * 32), 256, 0, stream>>>(qry, t2, g2d, b2d, qry, BQ);

    // flow head
    gemm(qry, nullptr, 0, F(MLP + 6), F(MLP + 7), t1, BQ, 256, 256, 1);
    gemm(t1, nullptr, 0, F(MLP + 8), F(MLP + 9), t2, BQ, 256, 256, 1);
    gemm(t2, nullptr, 0, F(MLP + 10), F(MLP + 11), fe, BQ, 2, 256, 0);
    flow_comb<<<elems(BQ * 2), 256, 0, stream>>>(refp, fe, flv, BQ * 2);

    // context head
    gemm(qry, nullptr, 0, F(MLP + 12), F(MLP + 13), t1, BQ, 256, 256, 1);
    gemm(t1, nullptr, 0, F(MLP + 14), F(MLP + 15), t2, BQ, 256, 256, 1);
    gemm(t2, nullptr, 0, F(MLP + 16), F(MLP + 17), ctxv, BQ, 256, 256, 0);

    // readout + convex upsample
    ctx_attn<<<elems((long)BATCH * H0 * W0 * 32), 256, 0, stream>>>(U1, ctxv, flv, cf);
    upsample_k<<<elems((long)BATCH * 2 * 8 * H0 * 8 * W0), 256, 0, stream>>>(
        cf, up_mask, (float*)d_out);
}
